// GraphAttentionLayer_49151605735876
// MI455X (gfx1250) — matrix operand order repaired and checked by reading, not compiled, not run on hardware
//
#include <hip/hip_runtime.h>
#include <hip/hip_bf16.h>
#include <math.h>

#define GAT_N    8192
#define GAT_FIN  512
#define GAT_FOUT 64
#define GAT_ALPHA 0.2f

typedef __attribute__((ext_vector_type(16))) __bf16 v16bf;
typedef __attribute__((ext_vector_type(8)))  __bf16 v8bf;
typedef __attribute__((ext_vector_type(8)))  float  v8f;

__device__ __forceinline__ float lrelu(float x) {
    return x > 0.0f ? x : GAT_ALPHA * x;
}

// ---------------------------------------------------------------------------
// Kernel 1: WT[c][k] = bf16(W[k][c])   (64 x 512 transposed bf16 copy of W)
// ---------------------------------------------------------------------------
__global__ void gat_prep_wt(const float* __restrict__ W, __bf16* __restrict__ WT) {
    int idx = blockIdx.x * blockDim.x + threadIdx.x;      // 0 .. 32767
    if (idx >= GAT_FIN * GAT_FOUT) return;
    int k = idx >> 6;          // row of W   (0..511)
    int c = idx & 63;          // col of W   (0..63)
    WT[(size_t)c * GAT_FIN + k] = (__bf16)W[idx];
}

// ---------------------------------------------------------------------------
// Kernel 2: h = x @ W via v_wmma_f32_16x16x32_bf16.
// One block = 16 rows of h; 4 waves, wave w owns output columns [16w,16w+16).
// Result written transposed+bf16: hT[col][row]  (64 x 8192).
// ---------------------------------------------------------------------------
__global__ void gat_gemm_h(const float* __restrict__ x,
                           const __bf16* __restrict__ WT,
                           __bf16* __restrict__ hT) {
    const int rowbase = blockIdx.x * 16;
    const int wave = threadIdx.x >> 5;     // 0..3
    const int lane = threadIdx.x & 31;
    const int hi   = lane >> 4;            // 0/1
    const int ln   = lane & 15;
    const int colbase = wave * 16;

    v8f c = {};
    for (int k = 0; k < GAT_FIN; k += 32) {
        // A fragment: x[rowbase+ln][k + 16*hi + 0..15]
        const float4* xp = (const float4*)(x + (size_t)(rowbase + ln) * GAT_FIN + k + 16 * hi);
        float4 a0 = xp[0];
        float4 a1 = xp[1];
        float4 a2 = xp[2];
        float4 a3 = xp[3];
        v16bf A;
        A[0]  = (__bf16)a0.x; A[1]  = (__bf16)a0.y; A[2]  = (__bf16)a0.z; A[3]  = (__bf16)a0.w;
        A[4]  = (__bf16)a1.x; A[5]  = (__bf16)a1.y; A[6]  = (__bf16)a1.z; A[7]  = (__bf16)a1.w;
        A[8]  = (__bf16)a2.x; A[9]  = (__bf16)a2.y; A[10] = (__bf16)a2.z; A[11] = (__bf16)a2.w;
        A[12] = (__bf16)a3.x; A[13] = (__bf16)a3.y; A[14] = (__bf16)a3.z; A[15] = (__bf16)a3.w;
        // B fragment: WT[colbase+ln][k + 16*hi + 0..15]  (16 contiguous bf16 = 32B)
        const v16bf* wb = (const v16bf*)(WT + (size_t)(colbase + ln) * GAT_FIN + k + 16 * hi);
        v16bf B = *wb;
        c = __builtin_amdgcn_wmma_f32_16x16x32_bf16(false, A, false, B, (short)0, c,
                                                    false, false);
    }
    // C layout: lane (hi,ln), VGPR v  ->  M = v + 8*hi, N = ln.
    // Store transposed: hT[colbase+ln][rowbase + 8*hi + v], v=0..7 contiguous.
    v8bf o;
#pragma unroll
    for (int v = 0; v < 8; ++v) o[v] = (__bf16)c[v];
    *(v8bf*)(hT + (size_t)(colbase + ln) * GAT_N + rowbase + 8 * hi) = o;
}

// ---------------------------------------------------------------------------
// Kernel 3: f1[i] = sum_c h[i][c]*a1[c] ; f2[i] = sum_c h[i][c]*a2[c]
// ---------------------------------------------------------------------------
__global__ void gat_compute_f(const __bf16* __restrict__ hT,
                              const float* __restrict__ a,
                              float* __restrict__ f1, float* __restrict__ f2) {
    int i = blockIdx.x * blockDim.x + threadIdx.x;
    if (i >= GAT_N) return;
    float s1 = 0.f, s2 = 0.f;
#pragma unroll 8
    for (int c = 0; c < GAT_FOUT; ++c) {
        float h = (float)hT[(size_t)c * GAT_N + i];
        s1 += h * a[c];
        s2 += h * a[GAT_FOUT + c];
    }
    f1[i] = s1;
    f2[i] = s2;
}

// ---------------------------------------------------------------------------
// Kernel 4: fused softmax(e*adj) @ h + ELU, flash-style, 16 rows per block.
// 128 threads = 4 waves; wave w owns output columns [16w, 16w+16).
// Thread t handles row (t>>3), columns (t&7)*8 + 0..7 of each 64-wide tile.
// ---------------------------------------------------------------------------
__global__ void gat_main(const float* __restrict__ adj,
                         const __bf16* __restrict__ hT,
                         const float* __restrict__ f1,
                         const float* __restrict__ f2,
                         float* __restrict__ out) {
    __shared__ float sm[16];          // per-row max
    __shared__ float sl[16];          // per-row exp-sum
    __shared__ float pmx[16][8];      // partial maxes
    __shared__ float pls[16][8];      // partial sums
    __shared__ __align__(16) __bf16 ptile[16][64];   // P tile, row-major

    const int rowbase = blockIdx.x * 16;
    const int t = threadIdx.x;        // 0..127
    const int ri = t >> 3;            // row 0..15 (both passes)
    const int s8 = t & 7;             // 0..7
    const float f1i = f1[rowbase + ri];
    const float* arow = adj + (size_t)(rowbase + ri) * GAT_N;

    // ---- Pass 1: per-row online (max, sum) over s = lrelu(f1+f2)*adj ----
    {
        float m = -__builtin_inff(), l = 0.f;
        for (int j = s8 * 4; j < GAT_N; j += 32) {
            __builtin_prefetch(arow + j + 512, 0, 0);
            float4 av = *(const float4*)(arow + j);
            float4 fv = *(const float4*)(f2 + j);
            float ev[4];
            ev[0] = lrelu(f1i + fv.x) * av.x;
            ev[1] = lrelu(f1i + fv.y) * av.y;
            ev[2] = lrelu(f1i + fv.z) * av.z;
            ev[3] = lrelu(f1i + fv.w) * av.w;
#pragma unroll
            for (int u = 0; u < 4; ++u) {
                float e = ev[u];
                if (e > m) { l *= __expf(m - e); m = e; }
                l += __expf(e - m);
            }
        }
        pmx[ri][s8] = m;
        pls[ri][s8] = l;
    }
    __syncthreads();
    if (t < 16) {
        float M = pmx[t][0], L = pls[t][0];
#pragma unroll
        for (int q = 1; q < 8; ++q) {
            float mq = pmx[t][q], lq = pls[t][q];
            if (mq > M) { L = L * __expf(M - mq) + lq; M = mq; }
            else        { L += lq * __expf(mq - M); }
        }
        sm[t] = M;
        sl[t] = L;
    }
    __syncthreads();

    // ---- Pass 2: accumulate P @ h with WMMA over 64-column tiles ----
    const int wave = t >> 5;
    const int lane = t & 31;
    const int hi = lane >> 4;
    const int ln = lane & 15;
    const int cof = s8 * 8;                  // this thread's 8 columns in tile
    const float mi = sm[ri];
    const float* arow2 = arow + cof;
    const __bf16* hrow = hT + (size_t)(16 * wave + ln) * GAT_N;

    v8f c = {};
    for (int jb = 0; jb < GAT_N; jb += 64) {
        __builtin_prefetch(arow2 + jb + 64, 0, 0);
        // compute 8 probabilities: columns jb+cof .. jb+cof+7 of row ri
        float4 a0 = *(const float4*)(arow2 + jb);
        float4 a1 = *(const float4*)(arow2 + jb + 4);
        float4 g0 = *(const float4*)(f2 + jb + cof);
        float4 g1 = *(const float4*)(f2 + jb + cof + 4);
        v8bf pv;
        pv[0] = (__bf16)__expf(lrelu(f1i + g0.x) * a0.x - mi);
        pv[1] = (__bf16)__expf(lrelu(f1i + g0.y) * a0.y - mi);
        pv[2] = (__bf16)__expf(lrelu(f1i + g0.z) * a0.z - mi);
        pv[3] = (__bf16)__expf(lrelu(f1i + g0.w) * a0.w - mi);
        pv[4] = (__bf16)__expf(lrelu(f1i + g1.x) * a1.x - mi);
        pv[5] = (__bf16)__expf(lrelu(f1i + g1.y) * a1.y - mi);
        pv[6] = (__bf16)__expf(lrelu(f1i + g1.z) * a1.z - mi);
        pv[7] = (__bf16)__expf(lrelu(f1i + g1.w) * a1.w - mi);
        *(v8bf*)&ptile[ri][cof] = pv;        // one ds_store_b128
        __syncthreads();

#pragma unroll
        for (int kk = 0; kk < 64; kk += 32) {
            // A fragment from LDS: row=ln, K = kk+16*hi+0..15
            v8bf A0 = *(const v8bf*)&ptile[ln][kk + 16 * hi];
            v8bf A1 = *(const v8bf*)&ptile[ln][kk + 16 * hi + 8];
            v16bf A = __builtin_shufflevector(A0, A1,
                        0, 1, 2, 3, 4, 5, 6, 7, 8, 9, 10, 11, 12, 13, 14, 15);
            // B fragment: hT[16*wave+ln][jb + kk + 16*hi + 0..15] (32B contiguous)
            v16bf B = *(const v16bf*)(hrow + jb + kk + 16 * hi);
            c = __builtin_amdgcn_wmma_f32_16x16x32_bf16(false, A, false, B, (short)0, c,
                                                        false, false);
        }
        __syncthreads();
    }

    // ---- Epilogue: normalize by row sum, ELU, store ----
#pragma unroll
    for (int v = 0; v < 8; ++v) {
        int row = v + 8 * hi;
        float val = c[v] / sl[row];
        float o = val > 0.f ? val : (__expf(val) - 1.f);
        out[(size_t)(rowbase + row) * GAT_FOUT + 16 * wave + ln] = o;
    }
}

// ---------------------------------------------------------------------------
extern "C" void kernel_launch(void* const* d_in, const int* in_sizes, int n_in,
                              void* d_out, int out_size, void* d_ws, size_t ws_size,
                              hipStream_t stream) {
    const float* x   = (const float*)d_in[0];   // 8192 x 512
    const float* adj = (const float*)d_in[1];   // 8192 x 8192
    const float* W   = (const float*)d_in[2];   // 512 x 64
    const float* a   = (const float*)d_in[3];   // 128 x 1
    float* out = (float*)d_out;                 // 8192 x 64

    char* ws = (char*)d_ws;
    __bf16* WT = (__bf16*)ws;                              // 64*512*2   = 64 KB
    __bf16* hT = (__bf16*)(ws + 64 * 1024);                // 64*8192*2  = 1 MB
    float*  f1 = (float*)(ws + 64 * 1024 + 1024 * 1024);   // 32 KB
    float*  f2 = (float*)(ws + 64 * 1024 + 1024 * 1024 + 32 * 1024);

    gat_prep_wt<<<(GAT_FIN * GAT_FOUT + 255) / 256, 256, 0, stream>>>(W, WT);
    gat_gemm_h<<<GAT_N / 16, 128, 0, stream>>>(x, WT, hT);
    gat_compute_f<<<GAT_N / 256, 256, 0, stream>>>(hT, a, f1, f2);
    gat_main<<<GAT_N / 16, 128, 0, stream>>>(adj, hT, f1, f2, out);
}
